// CycleGNN_78228534329619
// MI455X (gfx1250) — compile-verified, matrix operand order checked by reading, not run on hardware
//
#include <hip/hip_runtime.h>
#include <hip/hip_bf16.h>

#define TOTAL    131072
#define NMAX     2048
#define NGRAPH   64
#define FDIM     512
#define NFEAT    64
#define HID      128
#define NSTEPS   8

typedef __attribute__((ext_vector_type(16))) _Float16 v16h;
typedef __attribute__((ext_vector_type(8)))  float    v8f;

// ---------------------------------------------------------------------------
// K0: xs = x_start
// ---------------------------------------------------------------------------
__global__ __launch_bounds__(256) void cgnn_init_xs(
    const float* __restrict__ x_start, float* __restrict__ xs)
{
    int i = blockIdx.x * 256 + threadIdx.x;
    xs[i] = x_start[i];
}

// ---------------------------------------------------------------------------
// K1: fused 2-layer MLP  pred = W2 . relu([nf, xs] @ W1 + b1) + b2
// WMMA handles the 64 feature dims (K=64 -> two 16x16x32 f16 ops, f32 accum);
// the xs column, bias, relu and the 128->1 contraction are a VALU epilogue.
// Block: 256 threads = 8 waves; block owns 16 node rows; wave w owns cols
// [16w, 16w+16).
// ---------------------------------------------------------------------------
__global__ __launch_bounds__(256) void cgnn_mlp_wmma(
    const float* __restrict__ nf, const float* __restrict__ W1,
    const float* __restrict__ b1, const float* __restrict__ W2,
    const float* __restrict__ b2, const float* __restrict__ xs,
    float* __restrict__ pred, float* __restrict__ preds_out, int step)
{
    const int tid  = threadIdx.x;
    const int wave = tid >> 5;
    const int lane = tid & 31;
    const int half = lane >> 4;          // 0: rows/K-group A, 1: other half
    const int l15  = lane & 15;
    const int koff = half * 8;
    const int r0   = blockIdx.x * 16;
    const int c0   = wave * 16;

    __shared__ float xsh[16];
    __shared__ float wsum[8][16];
    if (tid < 16) xsh[tid] = xs[r0 + tid];
    __syncthreads();

    const int row = r0 + l15;
    const int col = c0 + l15;

    v8f acc = {};
    #pragma unroll
    for (int k0 = 0; k0 < NFEAT; k0 += 32) {
        // A fragment: lane (half,l15) holds row r0+l15, K = {koff..koff+7, koff+16..koff+23}
        const float* ab = nf + (size_t)row * NFEAT + k0 + koff;
        v16h afrag, bfrag;
        #pragma unroll
        for (int e = 0; e < 8; ++e) {
            afrag[e]     = (_Float16)ab[e];
            afrag[e + 8] = (_Float16)ab[e + 16];
        }
        // B fragment: lane holds col c0+l15, same K striping; W1 row-major [65][128]
        #pragma unroll
        for (int e = 0; e < 16; ++e) {
            int k = k0 + koff + (e < 8 ? e : e + 8);
            bfrag[e] = (_Float16)W1[k * HID + col];
        }
        acc = __builtin_amdgcn_wmma_f32_16x16x32_f16(
                  false, afrag, false, bfrag, (short)0, acc, false, false);
    }

    // Epilogue: + xs*W1[64,:], +b1, relu, *W2
    const float w1x  = W1[NFEAT * HID + col];
    const float bias = b1[col];
    const float w2c  = W2[col];
    float part[8];
    #pragma unroll
    for (int j = 0; j < 8; ++j) {
        float h = acc[j] + xsh[j + half * 8] * w1x + bias;
        h = fmaxf(h, 0.0f);
        part[j] = h * w2c;
    }
    // reduce over the 16 columns this wave owns (xor masks <16 stay in half-wave)
    #pragma unroll
    for (int j = 0; j < 8; ++j) {
        float v = part[j];
        #pragma unroll
        for (int s = 1; s < 16; s <<= 1) v += __shfl_xor(v, s, 32);
        part[j] = v;
    }
    if (l15 == 0) {
        #pragma unroll
        for (int j = 0; j < 8; ++j) wsum[wave][j + half * 8] = part[j];
    }
    __syncthreads();
    if (tid < 16) {
        float t = b2[0];
        #pragma unroll
        for (int w = 0; w < 8; ++w) t += wsum[w][tid];
        pred[r0 + tid] = t;
        preds_out[(size_t)(r0 + tid) * NSTEPS + step] = t;
    }
}

// ---------------------------------------------------------------------------
// K2: per-graph L1 norms -> labels[:,step] and direction d; also zeroes df.
// One 1024-thread block per graph (2 nodes/thread).
// ---------------------------------------------------------------------------
__global__ __launch_bounds__(1024) void cgnn_graph_norm(
    const float* __restrict__ xsol, const float* __restrict__ xs,
    const float* __restrict__ pred, float* __restrict__ labels_out,
    float* __restrict__ dvec, float* __restrict__ df, float tau, int step)
{
    const int b    = blockIdx.x;
    const int tid  = threadIdx.x;
    const int base = b * NMAX;
    const int wave = tid >> 5, lane = tid & 31;

    float sA = 0.f, sP = 0.f;
    #pragma unroll
    for (int r = 0; r < 2; ++r) {
        int i = base + tid + r * 1024;
        sA += fabsf(xsol[i] - xs[i]);
        sP += fabsf(pred[i]);
    }
    #pragma unroll
    for (int s = 1; s < 32; s <<= 1) {
        sA += __shfl_xor(sA, s, 32);
        sP += __shfl_xor(sP, s, 32);
    }
    __shared__ float redA[32], redP[32];
    __shared__ float SA, SP;
    if (lane == 0) { redA[wave] = sA; redP[wave] = sP; }
    if (tid < FDIM) df[b * FDIM + tid] = 0.0f;   // reset df for this step
    __syncthreads();
    if (wave == 0) {
        float a = redA[lane], p = redP[lane];
        #pragma unroll
        for (int s = 1; s < 32; s <<= 1) {
            a += __shfl_xor(a, s, 32);
            p += __shfl_xor(p, s, 32);
        }
        if (lane == 0) { SA = fmaxf(a, 1e-8f); SP = fmaxf(p, 1e-8f); }
    }
    __syncthreads();
    const float invA = 1.0f / SA, invP = 1.0f / SP;
    #pragma unroll
    for (int r = 0; r < 2; ++r) {
        int i = base + tid + r * 1024;
        labels_out[(size_t)i * NSTEPS + step] = (xsol[i] - xs[i]) * invA;
        dvec[i] = pred[i] * invP + 3.0f * tau / (xs[i] + tau);
    }
}

// ---------------------------------------------------------------------------
// K3: df[b,f] = sum_n P[b,n,f] * d[b,n].  Streams P column-wise, d staged
// in LDS; grid (B, F/256, 8 n-splits), atomicAdd combine. Prefetch ahead.
// ---------------------------------------------------------------------------
__global__ __launch_bounds__(256) void cgnn_proj_df(
    const float* __restrict__ P, const float* __restrict__ dvec,
    float* __restrict__ df)
{
    const int b  = blockIdx.x;
    const int f  = blockIdx.y * 256 + threadIdx.x;
    const int n0 = blockIdx.z * (NMAX / 8);

    __shared__ float dsh[NMAX / 8];
    dsh[threadIdx.x] = dvec[b * NMAX + n0 + threadIdx.x];
    __syncthreads();

    const float* Pb = P + ((size_t)(b * NMAX + n0)) * FDIM + f;
    float acc = 0.f;
    int n = 0;
    for (; n < (NMAX / 8) - 8; ++n) {
        __builtin_prefetch(Pb + (size_t)(n + 8) * FDIM, 0, 0);
        acc += Pb[(size_t)n * FDIM] * dsh[n];
    }
    for (; n < NMAX / 8; ++n)
        acc += Pb[(size_t)n * FDIM] * dsh[n];
    atomicAdd(&df[b * FDIM + f], acc);
}

// ---------------------------------------------------------------------------
// K4: proj[i] = P[i,:] . df[graph(i),:].  One wave32 per node row, float4
// loads (16 B/lane, fully coalesced), shfl reduction.
// ---------------------------------------------------------------------------
__global__ __launch_bounds__(256) void cgnn_proj_back(
    const float* __restrict__ P, const float* __restrict__ df,
    float* __restrict__ proj)
{
    const int wave = threadIdx.x >> 5, lane = threadIdx.x & 31;
    const int row  = blockIdx.x * 8 + wave;
    const int b    = row >> 11;

    const float4* Pr  = (const float4*)(P + (size_t)row * FDIM);
    const float4* Db  = (const float4*)(df + b * FDIM);
    float acc = 0.f;
    #pragma unroll
    for (int f4 = lane; f4 < FDIM / 4; f4 += 32) {
        float4 p = Pr[f4];
        float4 d = Db[f4];
        acc += p.x * d.x + p.y * d.y + p.z * d.z + p.w * d.w;
    }
    #pragma unroll
    for (int s = 1; s < 32; s <<= 1) acc += __shfl_xor(acc, s, 32);
    if (lane == 0) proj[row] = acc;
}

// ---------------------------------------------------------------------------
// K5: per-graph line search: alpha[b] = 0.995 * min(5, min_{d<0} xs/(-d))
// ---------------------------------------------------------------------------
__global__ __launch_bounds__(1024) void cgnn_line_search(
    const float* __restrict__ xs, const float* __restrict__ proj,
    float* __restrict__ alpha)
{
    const int b    = blockIdx.x;
    const int tid  = threadIdx.x;
    const int base = b * NMAX;
    const int wave = tid >> 5, lane = tid & 31;

    float m = 5.0f;
    #pragma unroll
    for (int r = 0; r < 2; ++r) {
        int i = base + tid + r * 1024;
        float p = proj[i];
        if (p < 0.f) m = fminf(m, xs[i] / (-p));
    }
    #pragma unroll
    for (int s = 1; s < 32; s <<= 1) m = fminf(m, __shfl_xor(m, s, 32));
    __shared__ float red[32];
    if (lane == 0) red[wave] = m;
    __syncthreads();
    if (wave == 0) {
        float v = red[lane];
        #pragma unroll
        for (int s = 1; s < 32; s <<= 1) v = fminf(v, __shfl_xor(v, s, 32));
        if (lane == 0) alpha[b] = fminf(v, 5.0f) * 0.995f;
    }
}

// ---------------------------------------------------------------------------
// K6: xs += alpha[graph] * proj
// ---------------------------------------------------------------------------
__global__ __launch_bounds__(256) void cgnn_xs_update(
    float* __restrict__ xs, const float* __restrict__ proj,
    const float* __restrict__ alpha)
{
    int i = blockIdx.x * 256 + threadIdx.x;
    xs[i] += alpha[i >> 11] * proj[i];
}

// ---------------------------------------------------------------------------
extern "C" void kernel_launch(void* const* d_in, const int* in_sizes, int n_in,
                              void* d_out, int out_size, void* d_ws, size_t ws_size,
                              hipStream_t stream) {
    const float* x_start = (const float*)d_in[0];
    const float* x_sol   = (const float*)d_in[1];
    const float* nf      = (const float*)d_in[2];
    const float* P       = (const float*)d_in[3];
    const float* W1      = (const float*)d_in[4];
    const float* b1      = (const float*)d_in[5];
    const float* W2      = (const float*)d_in[6];
    const float* b2      = (const float*)d_in[7];
    // d_in[8] (vals_batch) unused: graphs are dense equal-sized -> graph = i >> 11

    float* preds_out  = (float*)d_out;                           // [TOTAL, 8]
    float* labels_out = (float*)d_out + (size_t)TOTAL * NSTEPS;  // [TOTAL, 8]

    char* ws = (char*)d_ws;
    float* xs    = (float*)ws;                 ws += (size_t)TOTAL * 4;
    float* pred  = (float*)ws;                 ws += (size_t)TOTAL * 4;
    float* dvec  = (float*)ws;                 ws += (size_t)TOTAL * 4;
    float* proj  = (float*)ws;                 ws += (size_t)TOTAL * 4;
    float* df    = (float*)ws;                 ws += (size_t)NGRAPH * FDIM * 4;
    float* alpha = (float*)ws;                 ws += (size_t)NGRAPH * 4;

    cgnn_init_xs<<<TOTAL / 256, 256, 0, stream>>>(x_start, xs);

    double tau = 0.01;
    for (int s = 0; s < NSTEPS; ++s) {
        cgnn_mlp_wmma<<<TOTAL / 16, 256, 0, stream>>>(
            nf, W1, b1, W2, b2, xs, pred, preds_out, s);
        cgnn_graph_norm<<<NGRAPH, 1024, 0, stream>>>(
            x_sol, xs, pred, labels_out, dvec, df, (float)tau, s);
        cgnn_proj_df<<<dim3(NGRAPH, FDIM / 256, 8), 256, 0, stream>>>(P, dvec, df);
        cgnn_proj_back<<<TOTAL / 8, 256, 0, stream>>>(P, df, proj);
        cgnn_line_search<<<NGRAPH, 1024, 0, stream>>>(xs, proj, alpha);
        cgnn_xs_update<<<TOTAL / 256, 256, 0, stream>>>(xs, proj, alpha);
        tau = fmax(tau * 0.5, 1e-5);
    }
}
